// Recycling_56727928046505
// MI455X (gfx1250) — matrix operand order repaired and checked by reading, not compiled, not run on hardware
//
#include <hip/hip_runtime.h>
#include <hip/hip_bf16.h>
#include <math.h>

#define L   1024
#define DP  128     // d_pair
#define DS  32      // d_state
#define DM  256     // d_msa
#define FANIN 100   // 36 + 2*32

typedef __attribute__((ext_vector_type(16))) _Float16 v16h;
typedef __attribute__((ext_vector_type(8)))  float    v8f;

// ---------------------------------------------------------------------------
// Kernel 1: Cb coords from xyz + layernorm(state) -> state_n (output #3)
// ---------------------------------------------------------------------------
__global__ void prep_kernel(const float* __restrict__ xyz,
                            const float* __restrict__ state,
                            const float* __restrict__ g_state,
                            const float* __restrict__ b_state,
                            float* __restrict__ Cb,
                            float* __restrict__ state_n_out)
{
    int i = blockIdx.x * blockDim.x + threadIdx.x;
    if (i >= L) return;
    const float* x = xyz + (size_t)i * 9;
    float Nx = x[0], Ny = x[1], Nz = x[2];
    float Ax = x[3], Ay = x[4], Az = x[5];   // Ca
    float Cx = x[6], Cy = x[7], Cz = x[8];
    float bx = Ax - Nx, by = Ay - Ny, bz = Az - Nz;
    float cx = Cx - Ax, cy = Cy - Ay, cz = Cz - Az;
    float ax = by * cz - bz * cy;
    float ay = bz * cx - bx * cz;
    float az = bx * cy - by * cx;
    Cb[i * 3 + 0] = -0.58273431f * ax + 0.56802827f * bx - 0.54067466f * cx + Ax;
    Cb[i * 3 + 1] = -0.58273431f * ay + 0.56802827f * by - 0.54067466f * cy + Ay;
    Cb[i * 3 + 2] = -0.58273431f * az + 0.56802827f * bz - 0.54067466f * cz + Az;

    // layernorm over 32 state channels
    const float* s = state + (size_t)i * DS;
    float sum = 0.f, sum2 = 0.f;
    #pragma unroll
    for (int k = 0; k < DS; ++k) { float v = s[k]; sum += v; sum2 += v * v; }
    float mean = sum * (1.0f / DS);
    float var  = sum2 * (1.0f / DS) - mean * mean;
    float rstd = rsqrtf(var + 1e-5f);
    #pragma unroll
    for (int k = 0; k < DS; ++k)
        state_n_out[(size_t)i * DS + k] = (s[k] - mean) * rstd * g_state[k] + b_state[k];
}

// ---------------------------------------------------------------------------
// Kernel 2: left/right projection tables.
//  lr2[i][p] = b_proj[p] + sum_k state_n[i,k] * W[p, 36+k]
//  rr [j][p] =             sum_k state_n[j,k] * W[p, 68+k]
// ---------------------------------------------------------------------------
__global__ void proj_lr_kernel(const float* __restrict__ state_n,
                               const float* __restrict__ W,
                               const float* __restrict__ b_proj,
                               float* __restrict__ lr2,
                               float* __restrict__ rr)
{
    int t = blockIdx.x * blockDim.x + threadIdx.x;
    if (t >= L * DP) return;
    int i = t >> 7;         // row index
    int p = t & 127;        // pair channel
    const float* sn   = state_n + (size_t)i * DS;
    const float* wrow = W + (size_t)p * FANIN;
    float accL = b_proj[p];
    float accR = 0.f;
    #pragma unroll
    for (int k = 0; k < DS; ++k) {
        float v = sn[k];
        accL += v * wrow[36 + k];
        accR += v * wrow[68 + k];
    }
    lr2[t] = accL;
    rr[t]  = accR;
}

// ---------------------------------------------------------------------------
// Kernel 3: layernorm(msa) -> msa_out (output #1). One row per 256-thr block.
// ---------------------------------------------------------------------------
__global__ void ln_msa_kernel(const float* __restrict__ msa,
                              const float* __restrict__ g,
                              const float* __restrict__ b,
                              float* __restrict__ out)
{
    __shared__ float red[DM];
    int row = blockIdx.x, tid = threadIdx.x;
    float v = msa[(size_t)row * DM + tid];
    red[tid] = v; __syncthreads();
    for (int s = DM / 2; s > 0; s >>= 1) { if (tid < s) red[tid] += red[tid + s]; __syncthreads(); }
    float mean = red[0] * (1.0f / DM);
    __syncthreads();
    float d = v - mean;
    red[tid] = d * d; __syncthreads();
    for (int s = DM / 2; s > 0; s >>= 1) { if (tid < s) red[tid] += red[tid + s]; __syncthreads(); }
    float var = red[0] * (1.0f / DM);
    float rstd = rsqrtf(var + 1e-5f);
    out[(size_t)row * DM + tid] = d * rstd * g[tid] + b[tid];
}

// ---------------------------------------------------------------------------
// Kernel 4 (main): per wave, one tile of 16 pairs (fixed i, j0..j0+15):
//   pair_out[i,j,:] = RBF(D(i,j)) @ W_d^T   (WMMA f16, K padded 36->64)
//                    + left[i,:] + right[j,:] + b_proj + layernorm(pair[i,j,:])
// 128 threads = 4 waves per block; 4 tiles per block.
// ---------------------------------------------------------------------------
__global__ __launch_bounds__(128)
void pair_kernel(const float* __restrict__ pair,
                 const float* __restrict__ W,
                 const float* __restrict__ Cb,
                 const float* __restrict__ lr2,
                 const float* __restrict__ rr,
                 const float* __restrict__ g_pair,
                 const float* __restrict__ b_pair,
                 float* __restrict__ out)
{
    __shared__ __align__(16) _Float16 sW[DP * 64];        // W_d^T f16, K padded to 64
    __shared__ __align__(16) float    sPair[4][16 * DP];  // staged pair rows per wave
    __shared__ float                  sStats[4][16][2];   // {mean, rstd} per row

    // ---- fill W_d in f16, [p][k] layout, zero-pad k>=36 ----
    for (int idx = threadIdx.x; idx < DP * 64; idx += 128) {
        int p = idx >> 6, k = idx & 63;
        sW[idx] = (_Float16)((k < 36) ? W[(size_t)p * FANIN + k] : 0.0f);
    }
    __syncthreads();

    const int w  = threadIdx.x >> 5;   // wave in block
    const int l  = threadIdx.x & 31;   // lane
    const int T  = blockIdx.x * 4 + w; // tile id, 65536 total
    const int i  = T >> 6;             // 64 j-tiles per i
    const int j0 = (T & 63) << 4;

    // ---- stage 16 pair rows into LDS + layernorm stats (2 lanes per row) ----
    {
        int mr = l >> 1, h = l & 1;
        const float4* src = (const float4*)(pair + ((size_t)i * L + (j0 + mr)) * DP + h * 64);
        float4* dst = (float4*)(&sPair[w][mr * DP + h * 64]);
        float s = 0.f, s2 = 0.f;
        #pragma unroll
        for (int k = 0; k < 16; ++k) {
            float4 v = src[k];
            dst[k] = v;
            s  += v.x + v.y + v.z + v.w;
            s2 += v.x * v.x + v.y * v.y + v.z * v.z + v.w * v.w;
        }
        s  += __shfl_xor(s, 1);
        s2 += __shfl_xor(s2, 1);
        float mean = s * (1.0f / DP);
        float var  = s2 * (1.0f / DP) - mean * mean;
        float rstd = rsqrtf(var + 1e-5f);
        if (h == 0) { sStats[w][mr][0] = mean; sStats[w][mr][1] = rstd; }
    }

    // ---- build A fragments: RBF feats directly in WMMA A layout ----
    // A 16x32 f16: lane M = l&15; lanes<16 own K {0..7,16..23}, lanes>=16 own {8..15,24..31}
    const int m  = l & 15;
    const int jA = j0 + m;
    float dx = Cb[i * 3 + 0] - Cb[jA * 3 + 0];
    float dy = Cb[i * 3 + 1] - Cb[jA * 3 + 1];
    float dz = Cb[i * 3 + 2] - Cb[jA * 3 + 2];
    float D  = sqrtf(fmaxf(dx * dx + dy * dy + dz * dz, 1e-12f));
    const int koff = (l >> 4) * 8;
    v16h a0, a1;
    #pragma unroll
    for (int e = 0; e < 16; ++e) {
        int k  = (e < 8) ? (koff + e) : (16 + koff + (e - 8));
        // D_MU[k] = 2 + k*20/35 ; 1/D_SIGMA = 36/20 = 1.8
        float t = (D - (2.0f + 0.57142857142857140f * (float)k)) * 1.8f;
        a0[e] = (_Float16)__expf(-t * t);
    }
    #pragma unroll
    for (int e = 0; e < 16; ++e) {
        int k = 32 + ((e < 8) ? (koff + e) : (16 + koff + (e - 8)));
        float f = 0.0f;
        if (k < 36) {
            float t = (D - (2.0f + 0.57142857142857140f * (float)k)) * 1.8f;
            f = __expf(-t * t);
        }
        a1[e] = (_Float16)f;
    }

    // ---- 8 N-tiles of 16 channels: 2 WMMAs each + fused epilogue ----
    const int col = l & 15;
    const int hi  = l >> 4;
    for (int t = 0; t < 8; ++t) {
        // B 32x16 f16 layout: lane col = N, lanes<16 hold K {0..7,16..23}, lanes>=16 K {8..15,24..31}
        const _Float16* bs = &sW[(t * 16 + col) * 64 + hi * 8];
        v16h b0, b1;
        ((float4*)&b0)[0] = *(const float4*)(bs);
        ((float4*)&b0)[1] = *(const float4*)(bs + 16);
        ((float4*)&b1)[0] = *(const float4*)(bs + 32);
        ((float4*)&b1)[1] = *(const float4*)(bs + 48);

        v8f acc = {0.f, 0.f, 0.f, 0.f, 0.f, 0.f, 0.f, 0.f};
        acc = __builtin_amdgcn_wmma_f32_16x16x32_f16(false, a0, false, b0,
                                                     (short)0, acc, false, false);
        acc = __builtin_amdgcn_wmma_f32_16x16x32_f16(false, a1, false, b1,
                                                     (short)0, acc, false, false);

        const int p  = t * 16 + col;
        float lrv = lr2[(size_t)i * DP + p];
        float gp  = g_pair[p];
        float bp  = b_pair[p];
        #pragma unroll
        for (int r = 0; r < 8; ++r) {
            int mr = r + 8 * hi;          // D layout: vgpr r, lanes>=16 hold M=8+r
            int j  = j0 + mr;
            float pv   = sPair[w][mr * DP + p];
            float mean = sStats[w][mr][0];
            float rstd = sStats[w][mr][1];
            float lnv  = (pv - mean) * rstd * gp + bp;
            float res  = acc[r] + lrv + rr[(size_t)j * DP + p] + lnv;
            out[((size_t)i * L + j) * DP + p] = res;
        }
    }
}

// ---------------------------------------------------------------------------
extern "C" void kernel_launch(void* const* d_in, const int* in_sizes, int n_in,
                              void* d_out, int out_size, void* d_ws, size_t ws_size,
                              hipStream_t stream) {
    // input order: seq, msa, pair, xyz, state, W, b_proj,
    //              g_state, b_state, g_pair, b_pair, g_msa, b_msa
    const float* msa     = (const float*)d_in[1];
    const float* pair    = (const float*)d_in[2];
    const float* xyz     = (const float*)d_in[3];
    const float* state   = (const float*)d_in[4];
    const float* W       = (const float*)d_in[5];
    const float* b_proj  = (const float*)d_in[6];
    const float* g_state = (const float*)d_in[7];
    const float* b_state = (const float*)d_in[8];
    const float* g_pair  = (const float*)d_in[9];
    const float* b_pair  = (const float*)d_in[10];
    const float* g_msa   = (const float*)d_in[11];
    const float* b_msa   = (const float*)d_in[12];

    float* out       = (float*)d_out;
    float* msa_out   = out;                                   // 1024*256
    float* pair_out  = out + (size_t)L * DM;                  // 1024*1024*128
    float* state_out = pair_out + (size_t)L * L * DP;         // 1024*32

    float* wsf = (float*)d_ws;
    float* Cb  = wsf;                 // 1024*3  (use 4096 slot)
    float* lr2 = wsf + 4096;          // 1024*128
    float* rr  = wsf + 4096 + L * DP; // 1024*128

    prep_kernel<<<(L + 255) / 256, 256, 0, stream>>>(xyz, state, g_state, b_state,
                                                     Cb, state_out);
    proj_lr_kernel<<<(L * DP + 255) / 256, 256, 0, stream>>>(state_out, W, b_proj,
                                                             lr2, rr);
    ln_msa_kernel<<<L, DM, 0, stream>>>(msa, g_msa, b_msa, msa_out);

    // 65536 wave-tiles, 4 waves (128 threads) per block
    pair_kernel<<<65536 / 4, 128, 0, stream>>>(pair, W, Cb, lr2, rr,
                                               g_pair, b_pair, pair_out);
}
